// NeuralNet_37615323578557
// MI455X (gfx1250) — matrix operand order repaired and checked by reading, not compiled, not run on hardware
//
#include <hip/hip_runtime.h>

typedef __attribute__((ext_vector_type(16))) _Float16 v16h;
typedef __attribute__((ext_vector_type(8)))  _Float16 v8h;
typedef __attribute__((ext_vector_type(2)))  _Float16 h2;
typedef __attribute__((ext_vector_type(8)))  float    v8f;

#define HID 128
#define TM1 255      // T-1
#define TSTEPS 256

union BFrag { h2 p[8]; v16h v; };

__device__ __forceinline__ v16h lds_frag16(const _Float16* p0, const _Float16* p1) {
    v8h lo = *(const v8h*)p0;
    v8h hi = *(const v8h*)p1;
    return __builtin_shufflevector(lo, hi,
        0, 1, 2, 3, 4, 5, 6, 7, 8, 9, 10, 11, 12, 13, 14, 15);
}

__device__ __forceinline__ h2 cvt_pk_h2(float a, float b) {
    return __builtin_bit_cast(h2, __builtin_amdgcn_cvt_pkrtz(a, b));
}

// Transposed formulation: activations are h^T (features = M rows, batch = N cols),
// weights are the A operand (staged in LDS, K permuted by sigma so that the f32 C/D
// output registers ARE the next layer's B fragments after packed f16 conversion).
// sigma: K-chunk jj covers n-tiles {jj, jj+4}: slots 0..7 -> 16*jj+0..7,
// 8..15 -> 16*(jj+4)+0..7, 16..23 -> 16*jj+8..15, 24..31 -> 16*(jj+4)+8..15.
// MODE: 0 = fwd (store at s+1), 1 = bwd (add at s), 2 = init (store at 0)
template <int MODE>
__global__ __launch_bounds__(128)
__attribute__((amdgpu_num_vgpr(256)))   // stay in the low VGPR file: no s_set_vgpr_msb
void mlp_wmma_kernel(const float* __restrict__ x,   // (B, 257, 2)
                     const float* __restrict__ t,   // (B, 1)
                     const float* __restrict__ W1,  // (IN, 128)
                     const float* __restrict__ B1,  // (128,)
                     const float* __restrict__ W2,  // (128, 128)
                     const float* __restrict__ B2,  // (128,)
                     const float* __restrict__ W3,  // (128, 2)
                     const float* __restrict__ B3,  // (2,)
                     float* __restrict__ out,       // (B, 256, 2)
                     int numPairs)                  // number of 32-row tiles
{
    constexpr int IN = (MODE == 2) ? 3 : 8;

    // A-operand weight fragments, pre-swizzled per-lane: [tile][chunk][lane][8 halves]
    __shared__ __align__(16) _Float16 w1a[8][2][32][8];     // W1^T (+bias row at K=8): 8 KB
    __shared__ __align__(16) _Float16 w2a[8][4][2][32][8];  // W2^T, sigma-permuted K: 32 KB
    __shared__ __align__(16) _Float16 w3a[4][2][32][8];     // W3^T, sigma-permuted K: 4 KB
    __shared__ __align__(16) _Float16 sb2h[128];

    const int tid  = threadIdx.x;
    const int lane = tid & 31;
    const int wave = tid >> 5;

    // ---- stage pre-swizzled f16 weight A-fragments (128 threads) ----
    for (int idx = tid; idx < 8 * 2 * 32 * 8; idx += 128) {
        const int e = idx & 7, l = (idx >> 3) & 31, ch = (idx >> 8) & 1, j = idx >> 9;
        const int K = ((l < 16) ? 0 : 16) + e + (ch ? 8 : 0);
        const int n = 16 * j + (l & 15);
        _Float16 v = (_Float16)0.0f;
        if (K < IN)      v = (_Float16)W1[K * HID + n];
        else if (K == 8) v = (_Float16)B1[n];          // bias via ones-feature at K=8
        (&w1a[0][0][0][0])[idx] = v;
    }
    for (int idx = tid; idx < 8 * 4 * 2 * 32 * 8; idx += 128) {
        const int e = idx & 7, l = (idx >> 3) & 31, ch = (idx >> 8) & 1;
        const int jj = (idx >> 9) & 3, i = idx >> 11;
        const int slot = ((l < 16) ? 0 : 16) + e + (ch ? 8 : 0);
        const int tile = (slot & 8) ? (jj + 4) : jj;
        const int n1 = 16 * tile + ((slot & 16) ? 8 : 0) + (slot & 7);   // sigma(K)
        const int n2 = 16 * i + (l & 15);
        (&w2a[0][0][0][0][0])[idx] = (_Float16)W2[n1 * HID + n2];
    }
    for (int idx = tid; idx < 4 * 2 * 32 * 8; idx += 128) {
        const int e = idx & 7, l = (idx >> 3) & 31, ch = (idx >> 8) & 1, jj = idx >> 9;
        const int slot = ((l < 16) ? 0 : 16) + e + (ch ? 8 : 0);
        const int tile = (slot & 8) ? (jj + 4) : jj;
        const int n2 = 16 * tile + ((slot & 16) ? 8 : 0) + (slot & 7);   // sigma(K)
        const int n3 = l & 15;
        (&w3a[0][0][0][0])[idx] = (n3 < 2) ? (_Float16)W3[n2 * 2 + n3] : (_Float16)0.0f;
    }
    sb2h[tid] = (_Float16)B2[tid];
    __syncthreads();

    const float b30 = B3[0], b31 = B3[1];
    const int hiHalf = (lane >= 16);
    const int mofs   = hiHalf ? 8 : 0;
    const h2 zero2 = {(_Float16)0.0f, (_Float16)0.0f};

    const int waveId     = blockIdx.x * 4 + wave;
    const int waveStride = gridDim.x * 4;

    for (int tp = waveId; tp < numPairs; tp += waveStride) {
        const int rowBase = tp * 32;

        // Opaque zero: makes weight-fragment LDS addresses loop-variant so the
        // compiler cannot hoist the ds_loads out of the loop (which would force
        // ~350 VGPRs of weights live -> scratch spills under the 256-VGPR cap).
        int ofs0 = 0;
        asm volatile("" : "+v"(ofs0));
        const int laneO = lane + ofs0;

        // ==== features -> B fragments (feat^T: K = feature, N = batch in lanes) ====
        // lane<16: batch col = lane, halves 0..7 = f0..7, half 8 = 1.0 (bias), rest 0.
        // lane>=16 holds K=16..31 -> all zero padding.
        v16h fB[2];
        #pragma unroll
        for (int r = 0; r < 2; ++r) {
            v16h a = {};
            if (lane < 16) {
                const int row = rowBase + r * 16 + lane;
                float f[8];
                if (MODE == 2) {
                    const float* xb = x + (size_t)row * 514;
                    f[0] = xb[0]; f[1] = xb[1]; f[2] = t[row];
                    f[3] = f[4] = f[5] = f[6] = f[7] = 0.0f;
                } else {
                    const int b = row / TM1;
                    const int s = row - b * TM1;          // 0..254
                    const float* xb = x + (size_t)b * 514;
                    const float x0a = xb[s * 2 + 0], x0b = xb[s * 2 + 1];
                    const float x1a = xb[s * 2 + 2], x1b = xb[s * 2 + 3];
                    const float pt0 = xb[512], pt1 = xb[513];
                    const float tb  = t[b];
                    if (MODE == 0) { // fwd: [x[s+1], t, x[s], (s+1)/T, path_t]
                        f[0] = x1a; f[1] = x1b; f[2] = tb; f[3] = x0a; f[4] = x0b;
                        f[5] = (float)(s + 1) * (1.0f / 256.0f); f[6] = pt0; f[7] = pt1;
                    } else {         // bwd: [x[s], t, x[s+1], s/T, path_t]
                        f[0] = x0a; f[1] = x0b; f[2] = tb; f[3] = x1a; f[4] = x1b;
                        f[5] = (float)s * (1.0f / 256.0f); f[6] = pt0; f[7] = pt1;
                    }
                }
                #pragma unroll
                for (int i = 0; i < 8; ++i) a[i] = (_Float16)f[i];
                a[8] = (_Float16)1.0f;                   // ones row -> bias1
            }
            fB[r] = a;
        }

        // ==== layer 1: h1^T tiles; epilogue packs straight into layer-2 B frags ====
        BFrag B2f[4][2];
        #pragma unroll
        for (int j = 0; j < 8; ++j) {
            const v16h aW = lds_frag16(&w1a[j][0][laneO][0], &w1a[j][1][laneO][0]);
            #pragma unroll
            for (int r = 0; r < 2; ++r) {
                v8f z = {};
                const v8f acc = __builtin_amdgcn_wmma_f32_16x16x32_f16(
                    false, aW, false, fB[r], (short)0, z, false, false);
                #pragma unroll
                for (int q = 0; q < 4; ++q) {
                    h2 p = cvt_pk_h2(acc[2 * q], acc[2 * q + 1]);
                    p = __builtin_elementwise_max(p, zero2);   // bias already folded
                    if (j < 4) B2f[j][r].p[q] = p;
                    else       B2f[j - 4][r].p[q + 4] = p;
                }
            }
        }

        // ==== layer 2: h2^T = W2^T h1^T; epilogue packs into layer-3 B frags ====
        BFrag B3f[4][2];
        #pragma unroll
        for (int i = 0; i < 8; ++i) {
            h2 bb[4];
            __builtin_memcpy(bb, &sb2h[16 * i + mofs], 8);     // ds_load_b64 bias pairs
            v8f acc[2];
            #pragma unroll
            for (int jj = 0; jj < 4; ++jj) {
                const v16h aW = lds_frag16(&w2a[i][jj][0][laneO][0], &w2a[i][jj][1][laneO][0]);
                #pragma unroll
                for (int r = 0; r < 2; ++r) {
                    v8f cin;
                    if (jj == 0) { v8f z = {}; cin = z; }
                    else         { cin = acc[r]; }
                    acc[r] = __builtin_amdgcn_wmma_f32_16x16x32_f16(
                        false, aW, false, B2f[jj][r].v, (short)0, cin, false, false);
                }
            }
            #pragma unroll
            for (int r = 0; r < 2; ++r)
                #pragma unroll
                for (int q = 0; q < 4; ++q) {
                    h2 p = cvt_pk_h2(acc[r][2 * q], acc[r][2 * q + 1]);
                    p = p + bb[q];
                    p = __builtin_elementwise_max(p, zero2);
                    if (i < 4) B3f[i][r].p[q] = p;
                    else       B3f[i - 4][r].p[q + 4] = p;
                }
        }

        // ==== layer 3: out^T = W3^T h2^T (rows 0,1 valid) ====
        v8f acc3[2];
        #pragma unroll
        for (int jj = 0; jj < 4; ++jj) {
            const v16h aW = lds_frag16(&w3a[jj][0][laneO][0], &w3a[jj][1][laneO][0]);
            #pragma unroll
            for (int r = 0; r < 2; ++r) {
                v8f cin;
                if (jj == 0) { v8f z = {}; cin = z; }
                else         { cin = acc3[r]; }
                acc3[r] = __builtin_amdgcn_wmma_f32_16x16x32_f16(
                    false, aW, false, B3f[jj][r].v, (short)0, cin, false, false);
            }
        }

        // ==== write back: lane = batch row, regs 0/1 = the two outputs ====
        if (lane < 16) {
            #pragma unroll
            for (int r = 0; r < 2; ++r) {
                const int row = rowBase + r * 16 + lane;
                const float v0 = acc3[r][0] + b30;
                const float v1 = acc3[r][1] + b31;
                if (MODE == 2) {
                    float2* po = (float2*)&out[(size_t)row * (TSTEPS * 2)];
                    *po = make_float2(v0, v1);
                } else {
                    const int b = row / TM1;
                    const int s = row - b * TM1;
                    if (MODE == 0) {
                        float2* po = (float2*)&out[((size_t)b * TSTEPS + (s + 1)) * 2];
                        *po = make_float2(v0, v1);
                    } else {
                        float2* po = (float2*)&out[((size_t)b * TSTEPS + s) * 2];
                        float2 o = *po;                       // fwd/init done (stream order)
                        o.x += v0; o.y += v1;
                        *po = o;
                    }
                }
            }
        }
    }
}

extern "C" void kernel_launch(void* const* d_in, const int* in_sizes, int n_in,
                              void* d_out, int out_size, void* d_ws, size_t ws_size,
                              hipStream_t stream) {
    (void)in_sizes; (void)n_in; (void)out_size; (void)d_ws; (void)ws_size;

    const float* x  = (const float*)d_in[0];
    const float* t  = (const float*)d_in[1];
    const float* fw_W1 = (const float*)d_in[2];  const float* fw_b1 = (const float*)d_in[3];
    const float* fw_W2 = (const float*)d_in[4];  const float* fw_b2 = (const float*)d_in[5];
    const float* fw_W3 = (const float*)d_in[6];  const float* fw_b3 = (const float*)d_in[7];
    const float* bw_W1 = (const float*)d_in[8];  const float* bw_b1 = (const float*)d_in[9];
    const float* bw_W2 = (const float*)d_in[10]; const float* bw_b2 = (const float*)d_in[11];
    const float* bw_W3 = (const float*)d_in[12]; const float* bw_b3 = (const float*)d_in[13];
    const float* in_W1 = (const float*)d_in[14]; const float* in_b1 = (const float*)d_in[15];
    const float* in_W2 = (const float*)d_in[16]; const float* in_b2 = (const float*)d_in[17];
    const float* in_W3 = (const float*)d_in[18]; const float* in_b3 = (const float*)d_in[19];

    float* out = (float*)d_out;

    const int numPairsFB = (4096 * TM1) / 32;  // 32640
    const int numPairsI  = 4096 / 32;          // 128
    dim3 blk(128);

    // 1) fwd: stores score[:, 1:, :]
    mlp_wmma_kernel<0><<<1024, blk, 0, stream>>>(
        x, t, fw_W1, fw_b1, fw_W2, fw_b2, fw_W3, fw_b3, out, numPairsFB);
    // 2) init: stores score[:, 0, :]
    mlp_wmma_kernel<2><<<32, blk, 0, stream>>>(
        x, t, in_W1, in_b1, in_W2, in_b2, in_W3, in_b3, out, numPairsI);
    // 3) bwd: adds into score[:, :255, :]
    mlp_wmma_kernel<1><<<1024, blk, 0, stream>>>(
        x, t, bw_W1, bw_b1, bw_W2, bw_b2, bw_W3, bw_b3, out, numPairsFB);
}
